// Mobius_48730698940788
// MI455X (gfx1250) — compile-verified, matrix operand order checked by reading, not run on hardware
//
#include <hip/hip_runtime.h>

// ---------------------------------------------------------------------------
// Mobius iterative MLP on MI455X (gfx1250)
//   100 x [1024,2048]x[2048,2048]^T GEMM + relu (masked freeze), scalar
//   threshold per iter, final [1024,2048]x[1024,2048]^T GEMM.
// Strategy: bf16 WMMA (v_wmma_f32_16x16x32_bf16) with f32 accumulation.
//   - SADDR-form global loads (SGPR base + 32-bit VGPR offset)
//   - copy-free pipeline with bounded pressure: 2 rotating B slots
//     (refill for stage s+2 right after stage s reads the slot) and
//     A ping-pong across a x2-unrolled K body. <=2 B + 2 A fragments in
//     flight -> waits are loadcnt<=N without accumulator spills.
//   - #pragma unroll 1 on the K loop so the compiler cannot re-unroll and
//     double live ranges (round-4 spilled accumulators that way)
//   - final 64-K chunk peeled so lookahead loads never go out of bounds
// ---------------------------------------------------------------------------

typedef __bf16 bf16_t;
typedef __attribute__((ext_vector_type(8)))  float  v8f;
typedef __attribute__((ext_vector_type(8)))  bf16_t v8bf;
typedef __attribute__((ext_vector_type(16))) bf16_t v16bf;

#define IN_F   2048
#define OUT_F  1024
#define BATCH  1024
#define MAX_ITER 100
#define OUT_THRESHOLD 0.5f

union Frag16 { v16bf v; v8bf h[2]; };

// Load one 16(M)x32(K) bf16 fragment given a uniform base pointer and a
// 32-bit per-lane BYTE offset (SADDR-form global loads). CDNA5 16-bit layout:
//   lane r (0-15):  row r, K = {0..7, 16..23}   -> bytes {off+0..15, off+32..47}
//   lane r+16:      row r, K = {8..15, 24..31}  -> off already skewed by +16B
__device__ __forceinline__ v16bf load_frag16x32(const bf16_t* __restrict__ base,
                                                unsigned off) {
  Frag16 f;
  f.h[0] = *reinterpret_cast<const v8bf*>(reinterpret_cast<const char*>(base) + off);
  f.h[1] = *reinterpret_cast<const v8bf*>(reinterpret_cast<const char*>(base) + off + 32u);
  return f.v;
}

__device__ __forceinline__ v8f wmma_bf16(v16bf a, v16bf b, v8f c) {
  return __builtin_amdgcn_wmma_f32_16x16x32_bf16(false, a, false, b, (short)0, c,
                                                 false, false);
}

// One wave computes a 32(M) x 64(N) tile of X[M,K] * W[N,K]^T, K-major both.
__device__ __forceinline__ void wave_gemm_32x64(const bf16_t* __restrict__ X,
                                                const bf16_t* __restrict__ W,
                                                int K, int m0, int n0,
                                                v8f acc[2][4]) {
  const int lane = threadIdx.x & 31;
  const int r = lane & 15;
  const int hi = lane >> 4;

#pragma unroll
  for (int mi = 0; mi < 2; ++mi)
#pragma unroll
    for (int ni = 0; ni < 4; ++ni)
#pragma unroll
      for (int j = 0; j < 8; ++j) acc[mi][ni][j] = 0.0f;

  // 32-bit byte offsets, one per fragment row group (lane skew hi*16 bytes).
  unsigned aOff0 = (unsigned)(((m0 + r) * K) * 2 + hi * 16);
  unsigned aOff1 = aOff0 + (unsigned)(16 * K * 2);
  unsigned bOff  = (unsigned)(((n0 + r) * K) * 2 + hi * 16);
  const unsigned nS = (unsigned)(16 * K * 2);   // B tile stride along N

  // ---- pipeline prologue ----
  v16bf aE0 = load_frag16x32(X, aOff0);         // A, even K-half
  v16bf aE1 = load_frag16x32(X, aOff1);
  v16bf b0  = load_frag16x32(W, bOff);          // B slot 0 (stage 0)
  v16bf b1  = load_frag16x32(W, bOff + nS);     // B slot 1 (stage 1)
  v16bf aO0, aO1;                               // A, odd K-half

  // ---- main loop: 64 K per iteration; 8 stages, slot = stage & 1 ----
#pragma unroll 1
  for (int k = 0; k < K - 64; k += 64) {
    // s0 (k, ni=0)
    acc[0][0] = wmma_bf16(aE0, b0, acc[0][0]);
    acc[1][0] = wmma_bf16(aE1, b0, acc[1][0]);
    b0 = load_frag16x32(W, bOff + 2u * nS);             // for s2
    // s1 (k, ni=1)
    acc[0][1] = wmma_bf16(aE0, b1, acc[0][1]);
    acc[1][1] = wmma_bf16(aE1, b1, acc[1][1]);
    b1 = load_frag16x32(W, bOff + 3u * nS);             // for s3
    // s2 (k, ni=2)
    acc[0][2] = wmma_bf16(aE0, b0, acc[0][2]);
    acc[1][2] = wmma_bf16(aE1, b0, acc[1][2]);
    b0  = load_frag16x32(W, bOff + 64u);                // for s4
    aO0 = load_frag16x32(X, aOff0 + 64u);               // A for odd half
    // s3 (k, ni=3)
    acc[0][3] = wmma_bf16(aE0, b1, acc[0][3]);
    acc[1][3] = wmma_bf16(aE1, b1, acc[1][3]);
    b1  = load_frag16x32(W, bOff + 64u + nS);           // for s5
    aO1 = load_frag16x32(X, aOff1 + 64u);
    // s4 (k+32, ni=0)
    acc[0][0] = wmma_bf16(aO0, b0, acc[0][0]);
    acc[1][0] = wmma_bf16(aO1, b0, acc[1][0]);
    b0 = load_frag16x32(W, bOff + 64u + 2u * nS);       // for s6
    // s5 (k+32, ni=1)
    acc[0][1] = wmma_bf16(aO0, b1, acc[0][1]);
    acc[1][1] = wmma_bf16(aO1, b1, acc[1][1]);
    b1 = load_frag16x32(W, bOff + 64u + 3u * nS);       // for s7
    // s6 (k+32, ni=2)
    acc[0][2] = wmma_bf16(aO0, b0, acc[0][2]);
    acc[1][2] = wmma_bf16(aO1, b0, acc[1][2]);
    b0  = load_frag16x32(W, bOff + 128u);               // next iter s0
    aE0 = load_frag16x32(X, aOff0 + 128u);              // A for next even half
    // s7 (k+32, ni=3)
    acc[0][3] = wmma_bf16(aO0, b1, acc[0][3]);
    acc[1][3] = wmma_bf16(aO1, b1, acc[1][3]);
    b1  = load_frag16x32(W, bOff + 128u + nS);          // next iter s1
    aE1 = load_frag16x32(X, aOff1 + 128u);
    aOff0 += 128u;
    aOff1 += 128u;
    bOff  += 128u;
  }

  // ---- peeled final 64 K (k = K-64): in-chunk loads only, no lookahead ----
  {
    acc[0][0] = wmma_bf16(aE0, b0, acc[0][0]);
    acc[1][0] = wmma_bf16(aE1, b0, acc[1][0]);
    b0 = load_frag16x32(W, bOff + 2u * nS);
    acc[0][1] = wmma_bf16(aE0, b1, acc[0][1]);
    acc[1][1] = wmma_bf16(aE1, b1, acc[1][1]);
    b1 = load_frag16x32(W, bOff + 3u * nS);
    acc[0][2] = wmma_bf16(aE0, b0, acc[0][2]);
    acc[1][2] = wmma_bf16(aE1, b0, acc[1][2]);
    b0  = load_frag16x32(W, bOff + 64u);
    aO0 = load_frag16x32(X, aOff0 + 64u);
    acc[0][3] = wmma_bf16(aE0, b1, acc[0][3]);
    acc[1][3] = wmma_bf16(aE1, b1, acc[1][3]);
    b1  = load_frag16x32(W, bOff + 64u + nS);
    aO1 = load_frag16x32(X, aOff1 + 64u);
    acc[0][0] = wmma_bf16(aO0, b0, acc[0][0]);
    acc[1][0] = wmma_bf16(aO1, b0, acc[1][0]);
    b0 = load_frag16x32(W, bOff + 64u + 2u * nS);
    acc[0][1] = wmma_bf16(aO0, b1, acc[0][1]);
    acc[1][1] = wmma_bf16(aO1, b1, acc[1][1]);
    b1 = load_frag16x32(W, bOff + 64u + 3u * nS);
    acc[0][2] = wmma_bf16(aO0, b0, acc[0][2]);
    acc[1][2] = wmma_bf16(aO1, b0, acc[1][2]);
    acc[0][3] = wmma_bf16(aO0, b1, acc[0][3]);
    acc[1][3] = wmma_bf16(aO1, b1, acc[1][3]);
  }
}

// x_out = stopped ? x_in : relu(x_in @ Wfc^T + bfc)   (all bf16 activations)
// grid: (IN_F/128, BATCH/64), block: 128 threads (4 waves, 2x2 wave grid)
__global__ void fc_iter_kernel(const bf16_t* __restrict__ xin,
                               const bf16_t* __restrict__ Wfc,
                               const float* __restrict__ bfc,
                               bf16_t* __restrict__ xout,
                               const int* __restrict__ flags) {
  const int tid = threadIdx.x;
  if (flags[0]) {  // frozen: passthrough copy of this block's 64x128 tile
    const size_t rowBase = (size_t)blockIdx.y * 64;
    const size_t colBase = (size_t)blockIdx.x * 128;
    for (int c = tid; c < (64 * 128) / 8; c += blockDim.x) {
      const int rr = c >> 4;
      const int cc = (c & 15) * 8;
      *reinterpret_cast<v8bf*>(xout + (rowBase + rr) * IN_F + colBase + cc) =
          *reinterpret_cast<const v8bf*>(xin + (rowBase + rr) * IN_F + colBase + cc);
    }
    return;
  }
  const int w = tid >> 5;
  const int wm = w & 1;
  const int wn = w >> 1;
  const int m0 = blockIdx.y * 64 + wm * 32;
  const int n0 = blockIdx.x * 128 + wn * 64;

  v8f acc[2][4];
  wave_gemm_32x64(xin, Wfc, IN_F, m0, n0, acc);

  const int lane = tid & 31;
  const int r = lane & 15;
  const int hi = lane >> 4;
#pragma unroll
  for (int ni = 0; ni < 4; ++ni) {
    const int n = n0 + ni * 16 + r;           // C/D: N = lane % 16
    const float bias = bfc[n];
#pragma unroll
    for (int mi = 0; mi < 2; ++mi)
#pragma unroll
      for (int j = 0; j < 8; ++j) {
        const int m = m0 + mi * 16 + hi * 8 + j;  // C/D: M = vgpr + 8*(lane>=16)
        float v = acc[mi][ni][j] + bias;
        v = v > 0.0f ? v : 0.0f;
        xout[(size_t)m * IN_F + n] = (bf16_t)v;
      }
  }
}

// out = x @ Wout^T + bout, f32 result to d_out
// grid: (OUT_F/128, BATCH/64), block 128
__global__ void out_kernel(const bf16_t* __restrict__ xin,
                           const bf16_t* __restrict__ Wout,
                           const float* __restrict__ bout,
                           float* __restrict__ out) {
  const int tid = threadIdx.x;
  const int w = tid >> 5;
  const int wm = w & 1;
  const int wn = w >> 1;
  const int m0 = blockIdx.y * 64 + wm * 32;
  const int n0 = blockIdx.x * 128 + wn * 64;

  v8f acc[2][4];
  wave_gemm_32x64(xin, Wout, IN_F, m0, n0, acc);

  const int lane = tid & 31;
  const int r = lane & 15;
  const int hi = lane >> 4;
#pragma unroll
  for (int ni = 0; ni < 4; ++ni) {
    const int n = n0 + ni * 16 + r;
    const float bias = bout[n];
#pragma unroll
    for (int mi = 0; mi < 2; ++mi)
#pragma unroll
      for (int j = 0; j < 8; ++j) {
        const int m = m0 + mi * 16 + hi * 8 + j;
        out[(size_t)m * OUT_F + n] = acc[mi][ni][j] + bias;
      }
  }
}

// t = x[0] . W_th + b_th ; update last_i then stopped (reference ordering)
__global__ void threshold_kernel(const bf16_t* __restrict__ xcur,
                                 const float* __restrict__ Wth,
                                 const float* __restrict__ bth,
                                 int* __restrict__ flags, int idx) {
  __shared__ float red[256];
  float s = 0.0f;
  for (int k = threadIdx.x; k < IN_F; k += 256)
    s += (float)xcur[k] * Wth[k];
  red[threadIdx.x] = s;
  __syncthreads();
  for (int off = 128; off > 0; off >>= 1) {
    if ((int)threadIdx.x < off) red[threadIdx.x] += red[threadIdx.x + off];
    __syncthreads();
  }
  if (threadIdx.x == 0) {
    if (!flags[0]) {
      flags[1] = idx;                       // last executed iteration index
      const float t = red[0] + bth[0];
      if (t > OUT_THRESHOLD) flags[0] = 1;  // stop AFTER recording last_i
    }
  }
}

__global__ void f32_to_bf16_kernel(const float* __restrict__ in,
                                   bf16_t* __restrict__ out, int n) {
  const int i = blockIdx.x * blockDim.x + threadIdx.x;
  if (i < n) out[i] = (bf16_t)in[i];
}

__global__ void init_flags_kernel(int* flags) { flags[0] = 0; flags[1] = 0; }

__global__ void write_last_kernel(const int* __restrict__ flags,
                                  float* __restrict__ dout) {
  reinterpret_cast<int*>(dout)[(size_t)BATCH * OUT_F] = flags[1];
}

extern "C" void kernel_launch(void* const* d_in, const int* in_sizes, int n_in,
                              void* d_out, int out_size, void* d_ws, size_t ws_size,
                              hipStream_t stream) {
  const float* x    = (const float*)d_in[0];
  const float* Wfc  = (const float*)d_in[1];
  const float* bfc  = (const float*)d_in[2];
  const float* Wth  = (const float*)d_in[3];
  const float* bth  = (const float*)d_in[4];
  const float* Wout = (const float*)d_in[5];
  const float* bout = (const float*)d_in[6];

  char* ws = (char*)d_ws;
  bf16_t* Wfc_b  = (bf16_t*)(ws);                        // 2048*2048*2 = 8 MB
  bf16_t* Wout_b = (bf16_t*)(ws + ((size_t)8  << 20));   // 1024*2048*2 = 4 MB
  bf16_t* x0     = (bf16_t*)(ws + ((size_t)12 << 20));   // 1024*2048*2 = 4 MB
  bf16_t* x1     = (bf16_t*)(ws + ((size_t)16 << 20));   // 4 MB
  int*    flags  = (int*)   (ws + ((size_t)20 << 20));   // [stopped, last_i]

  init_flags_kernel<<<1, 1, 0, stream>>>(flags);
  f32_to_bf16_kernel<<<(IN_F * IN_F + 255) / 256, 256, 0, stream>>>(Wfc, Wfc_b, IN_F * IN_F);
  f32_to_bf16_kernel<<<(OUT_F * IN_F + 255) / 256, 256, 0, stream>>>(Wout, Wout_b, OUT_F * IN_F);
  f32_to_bf16_kernel<<<(BATCH * IN_F + 255) / 256, 256, 0, stream>>>(x, x0, BATCH * IN_F);

  bf16_t* cur = x0;
  bf16_t* nxt = x1;
  const dim3 fcGrid(IN_F / 128, BATCH / 64);
  for (int it = 0; it < MAX_ITER; ++it) {
    fc_iter_kernel<<<fcGrid, 128, 0, stream>>>(cur, Wfc_b, bfc, nxt, flags);
    threshold_kernel<<<1, 256, 0, stream>>>(nxt, Wth, bth, flags, it);
    bf16_t* t = cur; cur = nxt; nxt = t;
  }

  const dim3 outGrid(OUT_F / 128, BATCH / 64);
  out_kernel<<<outGrid, 128, 0, stream>>>(cur, Wout_b, bout, (float*)d_out);
  write_last_kernel<<<1, 1, 0, stream>>>(flags, (float*)d_out);
}
